// P_sLSTM_52733608460773
// MI455X (gfx1250) — compile-verified, matrix operand order checked by reading, not run on hardware
//
#include <hip/hip_runtime.h>
#include <hip/hip_bf16.h>
#include <math.h>

// ---------------- problem constants ----------------
constexpr int Bb     = 32;
constexpr int Ll     = 128;
constexpr int Cc     = 21;
constexpr int PATCH  = 16;
constexpr int Ee     = 128;   // E == H
constexpr int NP     = 8;     // patches per series
constexpr int BC     = Bb * Cc;          // 672
constexpr int MROWS  = BC * NP;          // 5376
constexpr int G4H    = 4 * Ee;           // 512 gates
constexpr int PRED   = 96;

typedef __bf16 bf16;
typedef bf16  v16bf __attribute__((ext_vector_type(16)));
typedef float v8f   __attribute__((ext_vector_type(8)));

union FragU { v16bf v; uint4 u[2]; };

// Load one 16x32 (or 32x16-transposed) bf16 WMMA fragment: lane reads its row's
// two 16-byte chunks at k-offsets half*8 and 16+half*8 (CDNA5 ISA 7.12.2).
__device__ __forceinline__ v16bf frag_row(const bf16* row, int half) {
    FragU f;
    f.u[0] = *reinterpret_cast<const uint4*>(row + half * 8);
    f.u[1] = *reinterpret_cast<const uint4*>(row + 16 + half * 8);
    return f.v;
}
__device__ __forceinline__ v16bf frag_zero() {
    FragU f;
    f.u[0].x = f.u[0].y = f.u[0].z = f.u[0].w = 0u;
    f.u[1] = f.u[0];
    return f.v;
}
__device__ __forceinline__ float sigm(float x) { return 1.f / (1.f + expf(-x)); }

// ---------------- weight conversion ----------------
__global__ void __launch_bounds__(256) k_f32_to_bf16(const float* __restrict__ in,
                                                     bf16* __restrict__ out, int n) {
    int i = blockIdx.x * 256 + threadIdx.x;
    if (i < n) out[i] = (bf16)in[i];
}

// conv_W (E, E, K=3) -> 3 row-major (E x E) matrices: out[k][o][i] = W[o][i][k]
__global__ void __launch_bounds__(256) k_convw(const float* __restrict__ W,
                                               bf16* __restrict__ out) {
    int i = blockIdx.x * 256 + threadIdx.x;
    int total = 3 * Ee * Ee;
    if (i >= total) return;
    int k = i / (Ee * Ee);
    int rem = i - k * Ee * Ee;
    int o = rem / Ee, ci = rem - o * Ee;
    out[i] = (bf16)W[(o * Ee + ci) * 3 + k];
}

// ---------------- stage 1: patchify + embed + layernorm ----------------
__global__ void __launch_bounds__(128)
k_embed_ln(const float* __restrict__ x, const float* __restrict__ eW,
           const float* __restrict__ eb, const float* __restrict__ lg,
           const float* __restrict__ lb, float* __restrict__ xnF,
           bf16* __restrict__ xnB) {
    int row = blockIdx.x;            // 0..MROWS-1, row = bc*NP + n
    int bc = row >> 3, n = row & 7;
    int b = bc / Cc, c = bc - b * Cc;
    __shared__ float patch[PATCH];
    __shared__ float red[Ee];
    __shared__ float smean, svar;
    int e = threadIdx.x;
    if (e < PATCH) patch[e] = x[((size_t)b * Ll + (n * PATCH + e)) * Cc + c];
    __syncthreads();
    float acc = eb[e];
#pragma unroll
    for (int p = 0; p < PATCH; ++p) acc += patch[p] * eW[e * PATCH + p];
    red[e] = acc;
    __syncthreads();
    for (int s = 64; s > 0; s >>= 1) { if (e < s) red[e] += red[e + s]; __syncthreads(); }
    if (e == 0) smean = red[0] * (1.f / Ee);
    __syncthreads();
    float d = acc - smean;
    red[e] = d * d;
    __syncthreads();
    for (int s = 64; s > 0; s >>= 1) { if (e < s) red[e] += red[e + s]; __syncthreads(); }
    if (e == 0) svar = red[0] * (1.f / Ee);
    __syncthreads();
    float xv = d * rsqrtf(svar + 1e-5f) * lg[e] + lb[e];
    xnF[(size_t)row * Ee + e] = xv;
    xnB[(size_t)row * Ee + e] = (bf16)xv;
}

// ---------------- generic bf16 WMMA GEMM: Y = A @ W^T (+bias) (+resid) ----------------
// A: (M x K) bf16 row-major, W: (Nout x K) bf16 row-major.
// block (32,8): each wave computes one 16x16 tile; grid (M/16, ceil(Nout/128)).
__global__ void __launch_bounds__(256)
k_gemm(const bf16* __restrict__ A, int lda, const bf16* __restrict__ W, int ldw,
       const float* __restrict__ bias, const float* __restrict__ resid,
       float* __restrict__ outF, bf16* __restrict__ outB,
       int M, int Nout, int K) {
    int wave = threadIdx.y, lane = threadIdx.x;
    int m0 = blockIdx.x * 16;
    int n0 = (blockIdx.y * 8 + wave) * 16;
    if (m0 >= M || n0 >= Nout) return;
    int mrow = lane & 15, half = lane >> 4, nc = lane & 15;
    v8f acc;
    float bv = bias ? bias[n0 + nc] : 0.f;
#pragma unroll
    for (int r = 0; r < 8; ++r) acc[r] = bv;
    const bf16* arow = A + (size_t)(m0 + mrow) * lda;
    const bf16* wrow = W + (size_t)(n0 + nc) * ldw;
    for (int kk = 0; kk < K; kk += 32) {
        v16bf af = frag_row(arow + kk, half);
        v16bf bfr = frag_row(wrow + kk, half);
        acc = __builtin_amdgcn_wmma_f32_16x16x32_bf16(false, af, false, bfr,
                                                      (short)0, acc, false, false);
    }
#pragma unroll
    for (int r = 0; r < 8; ++r) {
        size_t row = (size_t)(m0 + r + 8 * half);
        int col = n0 + nc;
        float v = acc[r];
        if (resid) v += resid[row * Nout + col];
        if (outF) outF[row * Nout + col] = v;
        if (outB) outB[row * Nout + col] = (bf16)v;
    }
}

// ---------------- stage 2b: attention (ctx = softmax-rowsum * v) ----------------
__global__ void __launch_bounds__(64)
k_attn(const float* __restrict__ q, const float* __restrict__ k,
       const float* __restrict__ v, bf16* __restrict__ ctxB) {
    int bc = blockIdx.x;
    int t = threadIdx.x;             // 64 = HEADS*NP
    int h = t >> 3, n = t & 7;
    const float* qn = q + ((size_t)(bc * NP + n)) * Ee + h * 16;
    float s[NP], mx = -1e30f;
    const float scale = 1.f / sqrtf((float)Ee);
#pragma unroll
    for (int l = 0; l < NP; ++l) {
        const float* kl = k + ((size_t)(bc * NP + l)) * Ee + h * 16;
        float d = 0.f;
#pragma unroll
        for (int j = 0; j < 16; ++j) d += qn[j] * kl[j];
        s[l] = d * scale;
        mx = fmaxf(mx, s[l]);
    }
    float se = 0.f;
#pragma unroll
    for (int l = 0; l < NP; ++l) { s[l] = expf(s[l] - mx); se += s[l]; }
    float rowsum = 0.f;
#pragma unroll
    for (int l = 0; l < NP; ++l) rowsum += s[l] / se;   // == reference's einsum factor
    const float* vn = v + ((size_t)(bc * NP + n)) * Ee + h * 16;
    bf16* out = ctxB + ((size_t)(bc * NP + n)) * Ee + h * 16;
#pragma unroll
    for (int j = 0; j < 16; ++j) out[j] = (bf16)(vn[j] * rowsum);
}

// ---------------- stage 3: persistent 2-layer xLSTM, 64 sequential steps ----------------
// Each block owns 16 rows of BC; h/c live in LDS for all 64 steps. gx0 holds the
// precomputed x_t @ Wx0^T + b0 (input is identical across outer iterations).
__global__ void __launch_bounds__(256)
k_xlstm(const float* __restrict__ gx0,         // (MROWS, 512), rows (bc*NP + t)
        const bf16* __restrict__ Wh0,          // (512,128)
        const bf16* __restrict__ Wx1,          // (512,128)
        const bf16* __restrict__ Wh1,          // (512,128)
        const float* __restrict__ b1,          // (512)
        float* __restrict__ outF,              // (MROWS,128)
        bf16* __restrict__ outB) {
    __shared__ __align__(16) bf16 h0b[16 * Ee];
    __shared__ __align__(16) bf16 h1b[16 * Ee];
    __shared__ float c0[16 * Ee];
    __shared__ float c1[16 * Ee];
    __shared__ float gsh[16 * G4H];
    int wave = threadIdx.y, lane = threadIdx.x;
    int tid = wave * 32 + lane;
    int r0 = blockIdx.x * 16;       // base row in BC
    for (int i = tid; i < 16 * Ee; i += 256) {
        h0b[i] = (bf16)0.f; h1b[i] = (bf16)0.f; c0[i] = 0.f; c1[i] = 0.f;
    }
    __syncthreads();
    int mrow = lane & 15, half = lane >> 4, nc = lane & 15;

    for (int outer = 0; outer < NP; ++outer) {
        for (int t = 0; t < NP; ++t) {
            // ---- layer 0 gates: gx0[t] + h0 @ Wh0^T ----
            for (int w4 = 0; w4 < 4; ++w4) {
                int n0 = (wave * 4 + w4) * 16;
                v8f acc;
#pragma unroll
                for (int r = 0; r < 8; ++r)
                    acc[r] = gx0[((size_t)(r0 + r + 8 * half) * NP + t) * G4H + n0 + nc];
                const bf16* ar = h0b + mrow * Ee;
                const bf16* wr = Wh0 + (size_t)(n0 + nc) * Ee;
                for (int kk = 0; kk < Ee; kk += 32)
                    acc = __builtin_amdgcn_wmma_f32_16x16x32_bf16(false, frag_row(ar + kk, half),
                            false, frag_row(wr + kk, half), (short)0, acc, false, false);
#pragma unroll
                for (int r = 0; r < 8; ++r) gsh[(r + 8 * half) * G4H + n0 + nc] = acc[r];
            }
            __syncthreads();
            // ---- layer 0 elementwise ----
            for (int i = tid; i < 16 * Ee; i += 256) {
                int rr = i >> 7, hh = i & 127;
                float ig = gsh[rr * G4H + hh];
                float fg = gsh[rr * G4H + Ee + hh];
                float gg = gsh[rr * G4H + 2 * Ee + hh];
                float og = gsh[rr * G4H + 3 * Ee + hh];
                float cn = sigm(fg) * c0[i] + sigm(ig) * tanhf(gg);
                float hn = sigm(og) * tanhf(cn);
                c0[i] = cn;
                h0b[i] = (bf16)hn;
            }
            __syncthreads();
            // ---- layer 1 gates: b1 + h0_new @ Wx1^T + h1 @ Wh1^T ----
            for (int w4 = 0; w4 < 4; ++w4) {
                int n0 = (wave * 4 + w4) * 16;
                v8f acc;
                float bv = b1[n0 + nc];
#pragma unroll
                for (int r = 0; r < 8; ++r) acc[r] = bv;
                const bf16* ar0 = h0b + mrow * Ee;
                const bf16* wr0 = Wx1 + (size_t)(n0 + nc) * Ee;
                for (int kk = 0; kk < Ee; kk += 32)
                    acc = __builtin_amdgcn_wmma_f32_16x16x32_bf16(false, frag_row(ar0 + kk, half),
                            false, frag_row(wr0 + kk, half), (short)0, acc, false, false);
                const bf16* ar1 = h1b + mrow * Ee;
                const bf16* wr1 = Wh1 + (size_t)(n0 + nc) * Ee;
                for (int kk = 0; kk < Ee; kk += 32)
                    acc = __builtin_amdgcn_wmma_f32_16x16x32_bf16(false, frag_row(ar1 + kk, half),
                            false, frag_row(wr1 + kk, half), (short)0, acc, false, false);
#pragma unroll
                for (int r = 0; r < 8; ++r) gsh[(r + 8 * half) * G4H + n0 + nc] = acc[r];
            }
            __syncthreads();
            // ---- layer 1 elementwise (+ emit outputs on final outer pass) ----
            for (int i = tid; i < 16 * Ee; i += 256) {
                int rr = i >> 7, hh = i & 127;
                float ig = gsh[rr * G4H + hh];
                float fg = gsh[rr * G4H + Ee + hh];
                float gg = gsh[rr * G4H + 2 * Ee + hh];
                float og = gsh[rr * G4H + 3 * Ee + hh];
                float cn = sigm(fg) * c1[i] + sigm(ig) * tanhf(gg);
                float hn = sigm(og) * tanhf(cn);
                c1[i] = cn;
                h1b[i] = (bf16)hn;
                if (outer == NP - 1) {
                    size_t row = (size_t)(r0 + rr) * NP + t;
                    outF[row * Ee + hh] = hn;
                    outB[row * Ee + hh] = (bf16)hn;
                }
            }
            __syncthreads();
        }
    }
}

// ---------------- stage 4: conv1d 'SAME' K=3 as 3 shifted WMMA GEMMs ----------------
__global__ void __launch_bounds__(256)
k_conv(const bf16* __restrict__ Xbf,   // (MROWS, E) lstm output
       const bf16* __restrict__ Wk,    // 3 * (E x E)
       const float* __restrict__ bias,
       bf16* __restrict__ outB) {
    int wave = threadIdx.y, lane = threadIdx.x;
    int m0 = blockIdx.x * 16;
    int n0 = wave * 16;
    int mrow = lane & 15, half = lane >> 4, nc = lane & 15;
    v8f acc;
    float bv = bias[n0 + nc];
#pragma unroll
    for (int r = 0; r < 8; ++r) acc[r] = bv;
    int n_in_seq = mrow & 7;          // m0 is a multiple of 16, NP == 8
    for (int k = 0; k < 3; ++k) {
        int sh = n_in_seq + k - 1;
        bool valid = (sh >= 0) && (sh < NP);
        const bf16* arow = Xbf + (size_t)(m0 + mrow + k - 1) * Ee;
        const bf16* wrow = Wk + (size_t)k * Ee * Ee + (size_t)nc * Ee;
        for (int kk = 0; kk < Ee; kk += 32) {
            v16bf af = valid ? frag_row(arow + kk, half) : frag_zero();
            acc = __builtin_amdgcn_wmma_f32_16x16x32_bf16(false, af, false,
                    frag_row(wrow + kk, half), (short)0, acc, false, false);
        }
    }
#pragma unroll
    for (int r = 0; r < 8; ++r)
        outB[(size_t)(m0 + r + 8 * half) * Ee + n0 + nc] = (bf16)acc[r];
}

// ---------------- host launcher ----------------
extern "C" void kernel_launch(void* const* d_in, const int* in_sizes, int n_in,
                              void* d_out, int out_size, void* d_ws, size_t ws_size,
                              hipStream_t stream) {
    (void)in_sizes; (void)n_in; (void)out_size; (void)ws_size;
    const float* x      = (const float*)d_in[0];
    const float* embW   = (const float*)d_in[1];
    const float* embB   = (const float*)d_in[2];
    const float* lng    = (const float*)d_in[3];
    const float* lnb    = (const float*)d_in[4];
    const float* qW     = (const float*)d_in[5];
    const float* qb     = (const float*)d_in[6];
    const float* kW     = (const float*)d_in[7];
    const float* kb     = (const float*)d_in[8];
    const float* vW     = (const float*)d_in[9];
    const float* vb     = (const float*)d_in[10];
    const float* lstmWx = (const float*)d_in[11];  // (2, 512, 128)
    const float* lstmWh = (const float*)d_in[12];
    const float* lstmB  = (const float*)d_in[13];  // (2, 512)
    const float* convW  = (const float*)d_in[14];
    const float* convB  = (const float*)d_in[15];
    const float* ssmW   = (const float*)d_in[16];
    const float* ssmB   = (const float*)d_in[17];
    const float* outpW  = (const float*)d_in[18];
    const float* outpB  = (const float*)d_in[19];
    const float* projW  = (const float*)d_in[20];
    const float* projB  = (const float*)d_in[21];

    char* cur = (char*)d_ws;
    auto alloc = [&](size_t bytes) { char* p = cur; cur += (bytes + 255) & ~(size_t)255; return p; };
    const size_t MF = (size_t)MROWS * Ee;
    float* xn_f    = (float*)alloc(MF * 4);
    bf16*  xn_bf   = (bf16*) alloc(MF * 2);
    float* q_f     = (float*)alloc(MF * 4);
    float* k_f     = (float*)alloc(MF * 4);
    float* v_f     = (float*)alloc(MF * 4);
    bf16*  ctx_bf  = (bf16*) alloc(MF * 2);
    float* gx0     = (float*)alloc((size_t)MROWS * G4H * 4);
    float* lstm_f  = (float*)alloc(MF * 4);
    bf16*  lstm_bf = (bf16*) alloc(MF * 2);
    bf16*  conv_bf = (bf16*) alloc(MF * 2);
    bf16*  t1_bf   = (bf16*) alloc(MF * 2);
    bf16*  t2_bf   = (bf16*) alloc(MF * 2);
    bf16*  qW_bf   = (bf16*) alloc((size_t)Ee * Ee * 2);
    bf16*  kW_bf   = (bf16*) alloc((size_t)Ee * Ee * 2);
    bf16*  vW_bf   = (bf16*) alloc((size_t)Ee * Ee * 2);
    bf16*  Wx_bf   = (bf16*) alloc((size_t)2 * G4H * Ee * 2);
    bf16*  Wh_bf   = (bf16*) alloc((size_t)2 * G4H * Ee * 2);
    bf16*  ssm_bf  = (bf16*) alloc((size_t)Ee * Ee * 2);
    bf16*  outp_bf = (bf16*) alloc((size_t)Ee * Ee * 2);
    bf16*  proj_bf = (bf16*) alloc((size_t)PRED * Ee * 2);
    bf16*  convW_bf= (bf16*) alloc((size_t)3 * Ee * Ee * 2);

    auto cvt = [&](const float* in, bf16* out, int n) {
        k_f32_to_bf16<<<(n + 255) / 256, 256, 0, stream>>>(in, out, n);
    };
    cvt(qW, qW_bf, Ee * Ee);
    cvt(kW, kW_bf, Ee * Ee);
    cvt(vW, vW_bf, Ee * Ee);
    cvt(lstmWx, Wx_bf, 2 * G4H * Ee);
    cvt(lstmWh, Wh_bf, 2 * G4H * Ee);
    cvt(ssmW, ssm_bf, Ee * Ee);
    cvt(outpW, outp_bf, Ee * Ee);
    cvt(projW, proj_bf, PRED * Ee);
    k_convw<<<(3 * Ee * Ee + 255) / 256, 256, 0, stream>>>(convW, convW_bf);

    // stage 1: embed + LN
    k_embed_ln<<<MROWS, 128, 0, stream>>>(x, embW, embB, lng, lnb, xn_f, xn_bf);

    dim3 blk(32, 8);
    int gm = MROWS / 16;  // 336
    // stage 2: q, k, v projections
    k_gemm<<<dim3(gm, 1), blk, 0, stream>>>(xn_bf, Ee, qW_bf, Ee, qb, nullptr, q_f, nullptr, MROWS, Ee, Ee);
    k_gemm<<<dim3(gm, 1), blk, 0, stream>>>(xn_bf, Ee, kW_bf, Ee, kb, nullptr, k_f, nullptr, MROWS, Ee, Ee);
    k_gemm<<<dim3(gm, 1), blk, 0, stream>>>(xn_bf, Ee, vW_bf, Ee, vb, nullptr, v_f, nullptr, MROWS, Ee, Ee);
    k_attn<<<BC, 64, 0, stream>>>(q_f, k_f, v_f, ctx_bf);

    // stage 3: precompute input gates for layer 0, then persistent xLSTM
    k_gemm<<<dim3(gm, 4), blk, 0, stream>>>(ctx_bf, Ee, Wx_bf, Ee, lstmB, nullptr, gx0, nullptr, MROWS, G4H, Ee);
    k_xlstm<<<BC / 16, blk, 0, stream>>>(gx0, Wh_bf, Wx_bf + (size_t)G4H * Ee,
                                         Wh_bf + (size_t)G4H * Ee, lstmB + G4H,
                                         lstm_f, lstm_bf);

    // stage 4: conv1d + ssm + outp(+residual xn) + proj
    k_conv<<<gm, blk, 0, stream>>>(lstm_bf, convW_bf, convB, conv_bf);
    k_gemm<<<dim3(gm, 1), blk, 0, stream>>>(conv_bf, Ee, ssm_bf, Ee, ssmB, nullptr, nullptr, t1_bf, MROWS, Ee, Ee);
    k_gemm<<<dim3(gm, 1), blk, 0, stream>>>(t1_bf, Ee, outp_bf, Ee, outpB, xn_f, nullptr, t2_bf, MROWS, Ee, Ee);
    k_gemm<<<dim3(gm, 1), blk, 0, stream>>>(t2_bf, Ee, proj_bf, Ee, projB, nullptr, (float*)d_out, nullptr, MROWS, PRED, Ee);
}
